// MultiHeadAttention_23699629540083
// MI455X (gfx1250) — matrix operand order repaired and checked by reading, not compiled, not run on hardware
//
#include <hip/hip_runtime.h>

// MHA forward for B=1,S=4096,H=1024,NH=16,DH=64 on gfx1250 (wave32, WMMA, TDM).
// f32->f16 convert, 3x WMMA projection GEMMs (32x64 tile/wave, V transposed),
// fused flash-attention: one wave per (head, 16-query tile), K/V tiles staged
// into LDS double-buffers by the Tensor Data Mover, online softmax.

typedef __attribute__((ext_vector_type(16))) _Float16 v16h;
typedef __attribute__((ext_vector_type(8)))  _Float16 v8h;
typedef __attribute__((ext_vector_type(8)))  float    v8f;
typedef __attribute__((ext_vector_type(4)))  float    v4f;
typedef __attribute__((ext_vector_type(4)))  unsigned int v4u;
typedef __attribute__((ext_vector_type(8)))  int      v8i;
typedef __attribute__((ext_vector_type(4)))  int      v4i;

#define SQ    4096
#define HDIM  1024
#define NHEAD 16
#define DHEAD 64

__device__ __forceinline__ int lane_id() { return threadIdx.x & 31; }

__device__ __forceinline__ v8f zero8() {
  v8f z;
#pragma unroll
  for (int i = 0; i < 8; ++i) z[i] = 0.f;
  return z;
}

// A fragment (16x32 f16, MxK) row-major, ld in halves. Lane m: K{0..7,16..23};
// lane m+16: K{8..15,24..31}.
__device__ __forceinline__ v16h load_frag_a(const _Float16* base, int ld) {
  const int l = lane_id();
  const _Float16* r = base + (size_t)(l & 15) * (size_t)ld + ((l >> 4) << 3);
  v8h lo = *(const v8h*)(r);
  v8h hi = *(const v8h*)(r + 16);
  v16h out;
#pragma unroll
  for (int i = 0; i < 8; ++i) { out[i] = lo[i]; out[i + 8] = hi[i]; }
  return out;
}

// B fragment (32x16 f16, KxN), column n contiguous with leading dim ld
// (source stored [N,K] row-major). Lane n: K{0..7,16..23}; lane n+16: K{8..15,24..31}.
__device__ __forceinline__ v16h load_frag_b(const _Float16* base, int ld) {
  const int l = lane_id();
  const _Float16* r = base + (size_t)(l & 15) * (size_t)ld + ((l >> 4) << 3);
  v8h lo = *(const v8h*)(r);
  v8h hi = *(const v8h*)(r + 16);
  v16h out;
#pragma unroll
  for (int i = 0; i < 8; ++i) { out[i] = lo[i]; out[i + 8] = hi[i]; }
  return out;
}

__device__ __forceinline__ v8f wmma16(v16h a, v16h b, v8f c) {
  return __builtin_amdgcn_wmma_f32_16x16x32_f16(false, a, false, b, (short)0, c,
                                                false, false);
}

__device__ __forceinline__ float redmax16(float x) {
#pragma unroll
  for (int m = 1; m < 16; m <<= 1) x = fmaxf(x, __shfl_xor(x, m, 16));
  return x;
}
__device__ __forceinline__ float redsum16(float x) {
#pragma unroll
  for (int m = 1; m < 16; m <<= 1) x += __shfl_xor(x, m, 16);
  return x;
}

// LDS byte offset of a __shared__-derived pointer (AS3 offset == low 32 bits
// of the generic VA formed by addrspacecast(3->0)).
__device__ __forceinline__ unsigned lds_off(const void* p) {
  return (unsigned)(unsigned long long)(size_t)p;
}

// Issue one TDM 2D tile load: global (row-major, stride0 elems of 2B) -> LDS.
// D# per CDNA5 ISA ch.8: group0 {count=1, lds_addr, global_addr, type=2},
// group1 {data_size=2B, tensor_dim0/1, tile_dim0/1, tensor_dim0_stride}.
__device__ __forceinline__ void tdm_load_2d(unsigned lds, const void* gptr,
                                            unsigned td0, unsigned td1,
                                            unsigned tile0, unsigned tile1,
                                            unsigned stride0) {
  const unsigned long long ga = (unsigned long long)(size_t)gptr;
  v4u g0;
  g0[0] = 1u;                                   // count=1, user mode
  g0[1] = lds;                                  // lds_addr
  g0[2] = (unsigned)ga;                         // global_addr[31:0]
  g0[3] = ((unsigned)(ga >> 32) & 0x01FFFFFFu) | 0x80000000u;  // addr[56:32]|type=2
  v8i g1;
  g1[0] = (int)(1u << 16);                      // workgroup_mask=0, data_size=1 (2B)
  g1[1] = (int)((td0 & 0xFFFFu) << 16);         // tensor_dim0[15:0]
  g1[2] = (int)((td0 >> 16) | ((td1 & 0xFFFFu) << 16));  // td0[31:16], td1[15:0]
  g1[3] = (int)((td1 >> 16) | (tile0 << 16));   // td1[31:16], tile_dim0
  g1[4] = (int)tile1;                           // tile_dim1 (tile_dim2=0)
  g1[5] = (int)stride0;                         // tensor_dim0_stride[31:0]
  g1[6] = 0;
  g1[7] = 0;
  v4i z4;
  z4[0] = 0; z4[1] = 0; z4[2] = 0; z4[3] = 0;   // groups 2/3 unused (2D)
  v8i z8;
#pragma unroll
  for (int i = 0; i < 8; ++i) z8[i] = 0;
  __builtin_amdgcn_tensor_load_to_lds(g0, g1, z4, z4, z8, 0);
}

// ---------------------------------------------------------------- convert
__global__ void __launch_bounds__(256) cvt_f32_f16(const float* __restrict__ in,
                                                   _Float16* __restrict__ out,
                                                   int n8) {
  int i = blockIdx.x * blockDim.x + threadIdx.x;
  if (i >= n8) return;
  v4f a = ((const v4f*)in)[2 * i + 0];
  v4f b = ((const v4f*)in)[2 * i + 1];
  v8h o;
#pragma unroll
  for (int j = 0; j < 4; ++j) { o[j] = (_Float16)a[j]; o[j + 4] = (_Float16)b[j]; }
  ((v8h*)out)[i] = o;
}

// ------------------------------------------------- projection GEMM + bias + relu
// out = relu(X @ W^T + b). 32x64 tile per wave; fragment loads hoisted into a
// clause ahead of the 8-WMMA burst so waits amortize.
__global__ void __launch_bounds__(128) proj_gemm_relu(
    const _Float16* __restrict__ X, const _Float16* __restrict__ W,
    const float* __restrict__ bias, _Float16* __restrict__ out,
    int transpose_out) {
  const int wid = blockIdx.x * (blockDim.x >> 5) + (threadIdx.x >> 5);
  const int mt = wid >> 4;          // 0..127 (32 rows each)
  const int nb = (wid & 15) << 6;   // col base, step 64
  const int m0 = mt << 5;
  const int l = lane_id();
  const int half = l >> 4, li = l & 15;

  v8f acc[2][4];
#pragma unroll
  for (int m = 0; m < 2; ++m)
#pragma unroll
    for (int n = 0; n < 4; ++n) acc[m][n] = zero8();

  const _Float16* xrow = X + (size_t)m0 * HDIM;
  for (int k0 = 0; k0 < HDIM; k0 += 32) {
    if (k0 + 32 < HDIM) {
      __builtin_prefetch(xrow + (size_t)li * HDIM + k0 + 32, 0, 1);
      __builtin_prefetch(W + (size_t)(nb + li) * HDIM + k0 + 32, 0, 1);
    }
    v16h a0 = load_frag_a(xrow + k0, HDIM);
    v16h a1 = load_frag_a(xrow + (size_t)16 * HDIM + k0, HDIM);
    v16h b[4];
#pragma unroll
    for (int n = 0; n < 4; ++n)
      b[n] = load_frag_b(W + (size_t)(nb + n * 16) * HDIM + k0, HDIM);
#pragma unroll
    for (int n = 0; n < 4; ++n) {
      acc[0][n] = wmma16(a0, b[n], acc[0][n]);
      acc[1][n] = wmma16(a1, b[n], acc[1][n]);
    }
  }

#pragma unroll
  for (int m = 0; m < 2; ++m) {
    const int mb = m0 + m * 16;
#pragma unroll
    for (int n = 0; n < 4; ++n) {
      const int col = nb + n * 16 + li;
      const float bv = bias[col];
      if (!transpose_out) {
#pragma unroll
        for (int v = 0; v < 8; ++v) {
          const int row = mb + v + half * 8;
          float x = acc[m][n][v] + bv;
          x = x > 0.f ? x : 0.f;
          out[(size_t)row * HDIM + col] = (_Float16)x;
        }
      } else {
        v8h pack;
#pragma unroll
        for (int v = 0; v < 8; ++v) {
          float x = acc[m][n][v] + bv;
          x = x > 0.f ? x : 0.f;
          pack[v] = (_Float16)x;
        }
        *(v8h*)(out + (size_t)col * SQ + mb + half * 8) = pack;
      }
    }
  }
}

// ------------------------------------------------------- fused flash attention
// One wave per workgroup (all TDM descriptor inputs blockIdx-uniform -> SGPRs).
// TDM double-buffers K (32x64) and V^T (64x32) tiles in LDS; online softmax;
// P transposed C->A layout through LDS.
__global__ void __launch_bounds__(32) attn_fused(
    const _Float16* __restrict__ Q, const _Float16* __restrict__ K,
    const _Float16* __restrict__ VT, const float* __restrict__ queries,
    float* __restrict__ out) {
  __shared__ __align__(16) _Float16 Kbuf[2][32][64];  // 8 KB
  __shared__ __align__(16) _Float16 Vbuf[2][64][32];  // 8 KB
  __shared__ __align__(16) _Float16 Pbuf[16][32];     // 1 KB

  const int wid = blockIdx.x;
  const int head = wid >> 8;   // 0..15
  const int qt = wid & 255;    // 0..255
  const int q0 = qt << 4;
  const int l = lane_id(), half = l >> 4, li = l & 15;

  const _Float16* kbase = K + head * DHEAD;                 // + kb*HDIM per tile
  const _Float16* vbase = VT + (size_t)(head * DHEAD) * SQ; // + kb per tile

  // prologue: TDM tiles for kb=0 into buffer 0
  tdm_load_2d(lds_off(&Kbuf[0][0][0]), kbase, HDIM, SQ, 64, 32, HDIM);
  tdm_load_2d(lds_off(&Vbuf[0][0][0]), vbase, SQ, HDIM, 32, 64, SQ);

  const _Float16* qbase = Q + (size_t)q0 * HDIM + head * DHEAD;
  const v16h aq0 = load_frag_a(qbase, HDIM);
  const v16h aq1 = load_frag_a(qbase + 32, HDIM);

  float mrow[8], lrow[8], sf[8];
  v8f o[4] = {zero8(), zero8(), zero8(), zero8()};
#pragma unroll
  for (int v = 0; v < 8; ++v) { mrow[v] = -1e30f; lrow[v] = 0.f; }

  for (int kb = 0; kb < SQ; kb += 32) {
    const int cur = (kb >> 5) & 1;
    if (kb + 32 < SQ) {  // prefetch next pair via TDM, wait current pair only
      tdm_load_2d(lds_off(&Kbuf[cur ^ 1][0][0]), kbase + (size_t)(kb + 32) * HDIM,
                  HDIM, SQ, 64, 32, HDIM);
      tdm_load_2d(lds_off(&Vbuf[cur ^ 1][0][0]), vbase + kb + 32,
                  SQ, HDIM, 32, 64, SQ);
      __builtin_amdgcn_s_wait_tensorcnt(2);
    } else {
      __builtin_amdgcn_s_wait_tensorcnt(0);
    }
    asm volatile("" ::: "memory");  // keep LDS reads below the tensorcnt wait

    // scores: 16 queries x 32 keys (K-dim 64 = 2x32), from LDS K tile
    const _Float16* kt = &Kbuf[cur][0][0];
    v8f s0 = wmma16(aq0, load_frag_b(kt, 64), zero8());
    s0 = wmma16(aq1, load_frag_b(kt + 32, 64), s0);
    v8f s1 = wmma16(aq0, load_frag_b(kt + 16 * 64, 64), zero8());
    s1 = wmma16(aq1, load_frag_b(kt + 16 * 64 + 32, 64), s1);

    // online softmax over 32 new columns (rows = v + 8*half)
#pragma unroll
    for (int v = 0; v < 8; ++v) {
      const float a0 = s0[v] * 0.125f;  // 1/sqrt(64)
      const float a1 = s1[v] * 0.125f;
      const float tmax = redmax16(fmaxf(a0, a1));
      const float mn = fmaxf(mrow[v], tmax);
      sf[v] = __expf(mrow[v] - mn);
      mrow[v] = mn;
      const float p0 = __expf(a0 - mn);
      const float p1 = __expf(a1 - mn);
      lrow[v] = lrow[v] * sf[v] + redsum16(p0 + p1);
      const int row = v + 8 * half;
      Pbuf[row][li] = (_Float16)p0;       // C-layout -> A-layout via LDS
      Pbuf[row][li + 16] = (_Float16)p1;
    }
#pragma unroll
    for (int n = 0; n < 4; ++n)
#pragma unroll
      for (int v = 0; v < 8; ++v) o[n][v] *= sf[v];

    const v16h ap = load_frag_a(&Pbuf[0][0], 32);
#pragma unroll
    for (int n = 0; n < 4; ++n) {
      v16h bv = load_frag_b(&Vbuf[cur][n * 16][0], 32);
      o[n] = wmma16(ap, bv, o[n]);
    }
  }

  // epilogue: a[h][q][d] -> torch-faithful permute/reshape + residual
#pragma unroll
  for (int n = 0; n < 4; ++n) {
    const int d = n * 16 + li;
#pragma unroll
    for (int v = 0; v < 8; ++v) {
      const int qg = q0 + v + 8 * half;
      const float val = o[n][v] / lrow[v];
      const int orow = head * 256 + d * 4 + (qg >> 10);
      const int ocol = qg & 1023;
      const size_t idx = (size_t)orow * HDIM + ocol;
      out[idx] = val + queries[idx];
    }
  }
}

// ----------------------------------------------------------------- launcher
extern "C" void kernel_launch(void* const* d_in, const int* in_sizes, int n_in,
                              void* d_out, int out_size, void* d_ws, size_t ws_size,
                              hipStream_t stream) {
  const float* queries = (const float*)d_in[0];
  const float* keys    = (const float*)d_in[1];
  const float* values  = (const float*)d_in[2];
  const float* Wq_w = (const float*)d_in[3];
  const float* Wq_b = (const float*)d_in[4];
  const float* Wk_w = (const float*)d_in[5];
  const float* Wk_b = (const float*)d_in[6];
  const float* Wv_w = (const float*)d_in[7];
  const float* Wv_b = (const float*)d_in[8];
  float* out = (float*)d_out;

  char* ws = (char*)d_ws;
  const size_t IN16 = (size_t)SQ * HDIM * sizeof(_Float16);    // 8 MB
  const size_t W16  = (size_t)HDIM * HDIM * sizeof(_Float16);  // 2 MB
  _Float16* qx = (_Float16*)(ws + 0 * IN16);
  _Float16* kx = (_Float16*)(ws + 1 * IN16);
  _Float16* vx = (_Float16*)(ws + 2 * IN16);
  _Float16* wq = (_Float16*)(ws + 3 * IN16 + 0 * W16);
  _Float16* wk = (_Float16*)(ws + 3 * IN16 + 1 * W16);
  _Float16* wv = (_Float16*)(ws + 3 * IN16 + 2 * W16);
  _Float16* qp = (_Float16*)(ws + 3 * IN16 + 3 * W16);
  _Float16* kp = (_Float16*)(ws + 4 * IN16 + 3 * W16);
  _Float16* vT = (_Float16*)(ws + 5 * IN16 + 3 * W16);  // [HDIM, SQ]

  const int nIn8 = SQ * HDIM / 8;
  const int nW8  = HDIM * HDIM / 8;
  cvt_f32_f16<<<(nIn8 + 255) / 256, 256, 0, stream>>>(queries, qx, nIn8);
  cvt_f32_f16<<<(nIn8 + 255) / 256, 256, 0, stream>>>(keys,    kx, nIn8);
  cvt_f32_f16<<<(nIn8 + 255) / 256, 256, 0, stream>>>(values,  vx, nIn8);
  cvt_f32_f16<<<(nW8 + 255) / 256, 256, 0, stream>>>(Wq_w, wq, nW8);
  cvt_f32_f16<<<(nW8 + 255) / 256, 256, 0, stream>>>(Wk_w, wk, nW8);
  cvt_f32_f16<<<(nW8 + 255) / 256, 256, 0, stream>>>(Wv_w, wv, nW8);

  const int projBlocks = (SQ / 32) * (HDIM / 64) / 4;  // 512 blocks x 4 waves
  proj_gemm_relu<<<projBlocks, 128, 0, stream>>>(qx, wq, Wq_b, qp, 0);
  proj_gemm_relu<<<projBlocks, 128, 0, stream>>>(kx, wk, Wk_b, kp, 0);
  proj_gemm_relu<<<projBlocks, 128, 0, stream>>>(vx, wv, Wv_b, vT, 1);

  const int attnBlocks = NHEAD * (SQ / 16);  // 4096 blocks x 1 wave
  attn_fused<<<attnBlocks, 32, 0, stream>>>(qp, kp, vT, queries, out);
}
